// GraphAttention_69320772157682
// MI455X (gfx1250) — matrix operand order repaired and checked by reading, not compiled, not run on hardware
//
#include <hip/hip_runtime.h>
#include <hip/hip_bf16.h>

// GAT layer: N=4096 nodes, F=512 in-feats, H=4 heads, FO=128 out-feats/head.
#define GN 4096
#define GF 512
#define GH 4
#define GFO 128

typedef __attribute__((ext_vector_type(16))) __bf16 v16bf;
typedef __attribute__((ext_vector_type(8)))  float  v8f;

__device__ __forceinline__ float lrelu_mask(float es, float en, float a) {
    float l = es + en;
    l = (l > 0.f) ? l : 0.2f * l;
    return l + (-1.0e10f) * (1.0f - a);
}

// =====================================================================
// Kernel 1: feats[h,n,o] = X[n,:] @ W[h,:,o]   (bf16 out, f32 accum)
// One wave per 16x64 output strip: A fragment loaded once per k-step,
// reused by 4 independent WMMAs (XDL pipelining, 4x less X traffic).
// Grid: (N/16, H*FO/64), Block: 32.
// =====================================================================
__global__ void feats_gemm(const float* __restrict__ X,
                           const float* __restrict__ W,
                           __bf16* __restrict__ feats) {
    const int lane = threadIdx.x;
    const int l15  = lane & 15;
    const int half = lane >> 4;

    const int m0 = blockIdx.x * 16;
    const int c0 = blockIdx.y * 64;        // 64-col strip (stays within one head)
    const int h  = c0 >> 7;
    const int ob = c0 & 127;

    const float* Xr = X + (size_t)(m0 + l15) * GF;
    const float* Wh = W + (size_t)h * GF * GFO;

    v8f acc[4];
    #pragma unroll
    for (int j = 0; j < 4; ++j) acc[j] = (v8f){0.f,0.f,0.f,0.f,0.f,0.f,0.f,0.f};

    for (int kb = 0; kb < GF; kb += 32) {
        // A fragment: 16x32 bf16 (lane<16: K 0..7 / 16..23; lane>=16: +8).
        v16bf af;
        #pragma unroll
        for (int v = 0; v < 8; ++v) {
            int kl = (v < 4) ? (half * 8 + 2 * v) : (16 + half * 8 + 2 * (v - 4));
            float2 p = *(const float2*)(Xr + kb + kl);
            af[2 * v]     = (__bf16)p.x;
            af[2 * v + 1] = (__bf16)p.y;
        }
        // 4 B fragments / 4 WMMAs (independent accumulators).
        #pragma unroll
        for (int j = 0; j < 4; ++j) {
            const int o = ob + j * 16 + l15;
            v16bf bf;
            #pragma unroll
            for (int v = 0; v < 8; ++v) {
                int f = kb + ((v < 4) ? (half * 8 + 2 * v) : (16 + half * 8 + 2 * (v - 4)));
                bf[2 * v]     = (__bf16)Wh[(size_t)f * GFO + o];
                bf[2 * v + 1] = (__bf16)Wh[(size_t)(f + 1) * GFO + o];
            }
            acc[j] = __builtin_amdgcn_wmma_f32_16x16x32_bf16(false, af, false, bf,
                                                             (short)0, acc[j], false, false);
        }
    }

    __bf16* fh = feats + (size_t)h * GN * GFO;
    #pragma unroll
    for (int j = 0; j < 4; ++j) {
        const int o = ob + j * 16 + l15;
        #pragma unroll
        for (int v = 0; v < 8; ++v) {
            int r = v + half * 8;
            fh[(size_t)(m0 + r) * GFO + o] = (__bf16)acc[j][v];
        }
    }
}

// =====================================================================
// Kernel 2: e_s[h,n] = feats[h,n,:].a_self[h,:], e_n likewise.
// =====================================================================
__global__ void escore(const __bf16* __restrict__ feats,
                       const float* __restrict__ a_self,
                       const float* __restrict__ a_neigh,
                       float* __restrict__ e_s, float* __restrict__ e_n) {
    int idx = blockIdx.x * blockDim.x + threadIdx.x;   // h*N + n
    if (idx >= GH * GN) return;
    int h = idx >> 12;
    const __bf16* f = feats + (size_t)idx * GFO;
    const float* as = a_self  + h * GFO;
    const float* an = a_neigh + h * GFO;
    float s = 0.f, n = 0.f;
    #pragma unroll 4
    for (int o = 0; o < GFO; ++o) {
        float v = (float)f[o];
        s += v * as[o];
        n += v * an[o];
    }
    e_s[idx] = s;
    e_n[idx] = n;
}

// =====================================================================
// Kernel 3: per (head, 16-row tile): masked-softmax attention + WMMA
// aggregate, with double-buffered async global->LDS staging of feats.
// Grid: (N/16, H), Block: 256 (8 waves; wave w owns out cols [16w,16w+16)).
// Chunk = 64 nodes: fs[2] double buffer = 2*64*136*2B = 34 KB LDS.
// A and e_n streamed with b128 per-lane loads (the roofline-dominant path).
// =====================================================================
#define CHUNK   64
#define FSPAD   136      // 128 o + 8 pad (bf16), row = 272B (16B aligned)
#define PPAD    72       // 64 + 8 pad

__global__ void gat_attn(const float* __restrict__ A,
                         const float* __restrict__ e_s,
                         const float* __restrict__ e_n,
                         const __bf16* __restrict__ feats,
                         const float* __restrict__ bias,
                         float* __restrict__ out) {
    const int n0  = blockIdx.x * 16;
    const int h   = blockIdx.y;
    const int tid = threadIdx.x;

    __shared__ float  s_m[16][17];
    __shared__ float  s_s[16][17];
    __shared__ float  rowmax[16];
    __shared__ float  rowinv[16];
    __shared__ __bf16 Pb[16 * PPAD];            // P tile [16 x 64] bf16
    __shared__ __bf16 fs[2][CHUNK * FSPAD];     // staged feats [64 x 128] bf16 x2

    const float* enh = e_n + (size_t)h * GN;
    const __bf16* fh = feats + (size_t)h * GN * GFO;

    // ---------- Phase A: online softmax stats (max, sum) per row ----------
    {
        int r = tid >> 4;
        int c = tid & 15;
        float es = e_s[(size_t)h * GN + n0 + r];
        const float4* Ar4 = (const float4*)(A + (size_t)(n0 + r) * GN);
        const float4* en4 = (const float4*)enh;
        float m = -3.0e38f, s = 0.f;
        for (int i = c; i < GN / 4; i += 16) {
            float4 a4 = Ar4[i];
            float4 e4 = en4[i];
            float l0 = lrelu_mask(es, e4.x, a4.x);
            float l1 = lrelu_mask(es, e4.y, a4.y);
            float l2 = lrelu_mask(es, e4.z, a4.z);
            float l3 = lrelu_mask(es, e4.w, a4.w);
            float lm = fmaxf(fmaxf(l0, l1), fmaxf(l2, l3));
            float e  = __expf(l0 - lm) + __expf(l1 - lm) +
                       __expf(l2 - lm) + __expf(l3 - lm);
            if (lm > m) { s = s * __expf(m - lm) + e; m = lm; }
            else        { s += e * __expf(lm - m); }
        }
        s_m[r][c] = m;
        s_s[r][c] = s;
    }
    __syncthreads();
    if (tid < 16) {
        float M = -3.0e38f, S = 0.f;
        #pragma unroll
        for (int c = 0; c < 16; ++c) {
            float mm = s_m[tid][c], ss = s_s[tid][c];
            if (mm > M) { S = S * __expf(M - mm) + ss; M = mm; }
            else        { S += ss * __expf(mm - M); }
        }
        rowmax[tid] = M;
        rowinv[tid] = 1.0f / S;
    }
    __syncthreads();

    // ---------- Phase B: async-staged feats + P tiles -> WMMA ----------
    const int lane = tid & 31;
    const int l15  = lane & 15;
    const int half = lane >> 4;
    const int o0   = (tid >> 5) * 16;          // wave's output-column base

    // Producer mapping for P: 4 consecutive values in row (tid&15).
    const int prow = tid & 15;
    const int pm0  = (tid >> 4) * 4;
    const float pes  = e_s[(size_t)h * GN + n0 + prow];
    const float pmax = rowmax[prow];
    const float pinv = rowinv[prow];
    const float* ArP = A + (size_t)(n0 + prow) * GN;

    // Async staging mapping: 4 x 16B segments per thread per chunk.
    const int snode = tid >> 4;                 // base node (stride 16 per iter)
    const int so16  = tid & 15;                 // 16B segment within 256B row

    // Prologue: async-load chunk 0 into fs[0].
    #pragma unroll
    for (int it = 0; it < 4; ++it) {
        int node = it * 16 + snode;
        unsigned ldsa = (unsigned)(size_t)&fs[0][node * FSPAD + so16 * 8];
        const __bf16* g = fh + (size_t)node * GFO + so16 * 8;
        asm volatile("global_load_async_to_lds_b128 %0, %1, off"
                     :: "v"(ldsa), "v"(g) : "memory");
    }

    v8f acc = {0.f,0.f,0.f,0.f,0.f,0.f,0.f,0.f};
    const int NCHUNK = GN / CHUNK;              // 64

    for (int ci = 0; ci < NCHUNK; ++ci) {
        const int cur = ci & 1;
        const int mb  = ci * CHUNK;

        // Issue async load of next chunk into the other buffer (overlaps
        // with the P build below and with this chunk's WMMAs).
        if (ci + 1 < NCHUNK) {
            const int nxt = cur ^ 1;
            const int mbn = mb + CHUNK;
            #pragma unroll
            for (int it = 0; it < 4; ++it) {
                int node = it * 16 + snode;
                unsigned ldsa = (unsigned)(size_t)&fs[nxt][node * FSPAD + so16 * 8];
                const __bf16* g = fh + (size_t)(mbn + node) * GFO + so16 * 8;
                asm volatile("global_load_async_to_lds_b128 %0, %1, off"
                             :: "v"(ldsa), "v"(g) : "memory");
            }
        }

        // Build P[16][64] = exp(l - rowmax) * rowinv (bf16), mask folded in.
        {
            const int j0 = mb + pm0;
            float4 a4 = *(const float4*)(ArP + j0);
            float4 e4 = *(const float4*)(enh + j0);
            float p0 = __expf(lrelu_mask(pes, e4.x, a4.x) - pmax) * pinv;
            float p1 = __expf(lrelu_mask(pes, e4.y, a4.y) - pmax) * pinv;
            float p2 = __expf(lrelu_mask(pes, e4.z, a4.z) - pmax) * pinv;
            float p3 = __expf(lrelu_mask(pes, e4.w, a4.w) - pmax) * pinv;
            __bf16* pp = &Pb[prow * PPAD + pm0];
            pp[0] = (__bf16)p0;
            pp[1] = (__bf16)p1;
            pp[2] = (__bf16)p2;
            pp[3] = (__bf16)p3;
        }

        // Wait for chunk ci (allow chunk ci+1's 4 loads to stay in flight).
        if (ci + 1 < NCHUNK) {
            asm volatile("s_wait_asynccnt 0x4" ::: "memory");
        } else {
            asm volatile("s_wait_asynccnt 0x0" ::: "memory");
        }
        __syncthreads();

        // 2 WMMAs covering the 64-node chunk (K = 32 each).
        #pragma unroll
        for (int kc = 0; kc < 2; ++kc) {
            const int kb = kc * 32;
            v16bf af, bf;
            #pragma unroll
            for (int v = 0; v < 8; ++v) {
                int kl = (v < 4) ? (half * 8 + 2 * v) : (16 + half * 8 + 2 * (v - 4));
                af[2 * v]     = Pb[l15 * PPAD + kb + kl];
                af[2 * v + 1] = Pb[l15 * PPAD + kb + kl + 1];
            }
            #pragma unroll
            for (int v = 0; v < 8; ++v) {
                int nl = kb + ((v < 4) ? (half * 8 + 2 * v) : (16 + half * 8 + 2 * (v - 4)));   // node within chunk
                bf[2 * v]     = fs[cur][nl * FSPAD + o0 + l15];
                bf[2 * v + 1] = fs[cur][(nl + 1) * FSPAD + o0 + l15];
            }
            acc = __builtin_amdgcn_wmma_f32_16x16x32_bf16(false, af, false, bf,
                                                          (short)0, acc, false, false);
        }
        __syncthreads();
    }

    // Write out: out[n, h*FO + o], + bias, ReLU.
    const float bv = bias[h * GFO + o0 + l15];
    #pragma unroll
    for (int v = 0; v < 8; ++v) {
        int r = v + half * 8;
        float x = acc[v] + bv;
        out[(size_t)(n0 + r) * (GH * GFO) + h * GFO + o0 + l15] = (x > 0.f) ? x : 0.f;
    }
}

// =====================================================================
extern "C" void kernel_launch(void* const* d_in, const int* in_sizes, int n_in,
                              void* d_out, int out_size, void* d_ws, size_t ws_size,
                              hipStream_t stream) {
    const float* X       = (const float*)d_in[0];   // [N,F]
    const float* A       = (const float*)d_in[1];   // [N,N]
    const float* W       = (const float*)d_in[2];   // [H,F,FO]
    const float* b       = (const float*)d_in[3];   // [H,FO]
    const float* a_self  = (const float*)d_in[4];   // [H,FO]
    const float* a_neigh = (const float*)d_in[5];   // [H,FO]
    float* out = (float*)d_out;                     // [N, H*FO]

    // Workspace: feats bf16 [H,N,FO] (4 MB) + e_s [H,N] + e_n [H,N]
    __bf16* feats = (__bf16*)d_ws;
    float* e_s = (float*)((char*)d_ws + (size_t)GH * GN * GFO * sizeof(__bf16));
    float* e_n = e_s + (size_t)GH * GN;

    feats_gemm<<<dim3(GN / 16, (GH * GFO) / 64), 32, 0, stream>>>(X, W, feats);
    escore<<<(GH * GN + 255) / 256, 256, 0, stream>>>(feats, a_self, a_neigh, e_s, e_n);
    gat_attn<<<dim3(GN / 16, GH), 256, 0, stream>>>(A, e_s, e_n, feats, b, out);
}
